// LNet_69303592288588
// MI455X (gfx1250) — matrix operand order repaired and checked by reading, not compiled, not run on hardware
//
#include <hip/hip_runtime.h>
#include <hip/hip_bf16.h>
#include <stddef.h>
#include <stdint.h>

#define USE_TDM 1

// ---------------------------------------------------------------------------
// Types for CDNA5 WMMA (wave32): v_wmma_f32_16x16x32_bf16
// ---------------------------------------------------------------------------
typedef __bf16 bf16;
typedef __attribute__((ext_vector_type(16))) __bf16 v16bf;
typedef __attribute__((ext_vector_type(8)))  float  v8f;
typedef __attribute__((ext_vector_type(4)))  int    v4i;
typedef __attribute__((ext_vector_type(8)))  int    v8i;

#define WMMA_BF16(a, b, c) \
  __builtin_amdgcn_wmma_f32_16x16x32_bf16(false, (a), false, (b), (short)0, (c), false, false)

// ---------------------------------------------------------------------------
// WMMA tile loaders/stores per ISA 7.12.2 (wave32 layouts)
// ---------------------------------------------------------------------------
__device__ __forceinline__ v16bf load_A_tile(const bf16* A, int lda, int m0, int k0, int lane) {
  int m  = lane & 15;
  int kb = (lane < 16) ? 0 : 8;
  const bf16* p = A + (size_t)(m0 + m) * lda + k0 + kb;
  v16bf a;
#pragma unroll
  for (int e = 0; e < 8; ++e) { a[e] = p[e]; a[e + 8] = p[e + 16]; }
  return a;
}

__device__ __forceinline__ v16bf load_B_tile_nk(const bf16* Wnk, int ldk, int k0, int n0, int lane) {
  int n  = lane & 15;
  int kh = (lane < 16) ? 0 : 8;
  const bf16* p = Wnk + (size_t)(n0 + n) * ldk + k0 + kh;
  v16bf b;
#pragma unroll
  for (int e = 0; e < 8; ++e) { b[e] = p[e]; b[e + 8] = p[e + 16]; }
  return b;
}

__device__ __forceinline__ void store_D_f32(float* D, int ldn, int m0, int n0, int lane, v8f d) {
  int n = lane & 15, mh = (lane < 16) ? 0 : 8;
#pragma unroll
  for (int r = 0; r < 8; ++r) D[(size_t)(m0 + mh + r) * ldn + n0 + n] = d[r];
}

__device__ __forceinline__ float sigf(float x) { return 1.0f / (1.0f + __expf(-x)); }

__device__ __forceinline__ int   f2ord(float f) { int i = __float_as_int(f); return (i >= 0) ? i : (i ^ 0x7fffffff); }
__device__ __forceinline__ float ord2f(int k)   { int i = (k >= 0) ? k : (k ^ 0x7fffffff); return __int_as_float(i); }

// ---------------------------------------------------------------------------
// Tensor Data Mover: 2-D tile (bf16 elements) global -> LDS.
// D# per CDNA5 ISA: group0 {ctl, lds_addr, gaddr lo/hi|type},
// group1 {data_size=2B, tensor_dim0/1 == tile_dim0/1, dim0 stride}, groups 2/3
// benign (tensors up to 5D form so the 4-operand asm shape is used).
// Issued once per wave; caller restricts to wave 0 and waits on TENSORcnt.
// ---------------------------------------------------------------------------
#if USE_TDM
__device__ __forceinline__ void tdm_load_tile_bf16(uint64_t gaddr, uint32_t lds_off,
                                                   int tile_d0, int tile_d1,
                                                   int stride_elems) {
  v4i g0; v8i g1; v4i g2, g3;
  g0[0] = 1;                                                   // count=1 valid user D#
  g0[1] = (int)lds_off;                                        // lds_addr
  g0[2] = (int)(uint32_t)(gaddr & 0xffffffffull);              // global_addr[31:0]
  g0[3] = (int)((uint32_t)((gaddr >> 32) & 0x01ffffffull) | (2u << 30)); // addr[56:32] | type=2
  g1[0] = (1 << 16);                                           // data_size=1 (2 bytes), mask=0
  g1[1] = (tile_d0 & 0xffff) << 16;                            // tensor_dim0[15:0]
  g1[2] = ((unsigned)tile_d0 >> 16) | ((tile_d1 & 0xffff) << 16); // td0[31:16] | tensor_dim1[15:0]
  g1[3] = ((unsigned)tile_d1 >> 16) | ((tile_d0 & 0xffff) << 16); // td1[31:16] | tile_dim0
  g1[4] = (tile_d1 & 0xffff);                                  // tile_dim1 (tile_dim2=0)
  g1[5] = stride_elems;                                        // tensor_dim0_stride[31:0]
  g1[6] = 0;                                                   // stride hi | dim1_stride lo
  g1[7] = 0;
  g2[0] = 1; g2[1] = 1; g2[2] = 0; g2[3] = 0;                  // tensor_dim2/3 = 1
  g3[0] = 0; g3[1] = 0; g3[2] = 0; g3[3] = 0;
  asm volatile("tensor_load_to_lds %0, %1, %2, %3"
               :: "s"(g0), "s"(g1), "s"(g2), "s"(g3)
               : "memory");
}
#endif

// ---------------------------------------------------------------------------
// Setup kernels
// ---------------------------------------------------------------------------
__global__ void init_kernel(int* minmax, float* actions, bf16* dec_in) {
  int i = blockIdx.x * 256 + threadIdx.x;
  if (i == 0) { minmax[0] = 0x7fffffff; minmax[1] = 0x80000000; }
  if (i < 64 * 512) actions[i] = 0.0f;
  if (i < 64 * 256) dec_in[i] = (bf16)0.0f;
}

__global__ void cvt_bf16_kernel(const float* src, bf16* dst, int n) {
  int i = blockIdx.x * 256 + threadIdx.x;
  if (i < n) dst[i] = (bf16)src[i];
}

__global__ void emb_kernel(const float* __restrict__ s_inputs, const float* __restrict__ emb_W,
                           bf16* __restrict__ emb, int* minmax) {
  int gid = blockIdx.x * 256 + threadIdx.x;   // B*S*E = 4194304
  int e = gid & 127, bs = gid >> 7;
  float v = s_inputs[bs * 3 + 0] * emb_W[e * 3 + 0] +
            s_inputs[bs * 3 + 1] * emb_W[e * 3 + 1] +
            s_inputs[bs * 3 + 2] * emb_W[e * 3 + 2];
  emb[gid] = (bf16)v;
  __shared__ int rmin[256], rmax[256];
  int t = threadIdx.x, k = f2ord(v);
  rmin[t] = k; rmax[t] = k; __syncthreads();
  for (int s = 128; s > 0; s >>= 1) {
    if (t < s) { rmin[t] = min(rmin[t], rmin[t + s]); rmax[t] = max(rmax[t], rmax[t + s]); }
    __syncthreads();
  }
  if (t == 0) { atomicMin(&minmax[0], rmin[0]); atomicMax(&minmax[1], rmax[0]); }
}

__global__ void norm_emb_kernel(bf16* emb, const int* minmax) {
  int gid = blockIdx.x * 256 + threadIdx.x;
  float mn = ord2f(minmax[0]), mx = ord2f(minmax[1]);
  float inv = 1.0f / (mx - mn);
  emb[gid] = (bf16)(((float)emb[gid] - mn) * inv);
}

// ---------------------------------------------------------------------------
// Encoder: persistent 512-step LSTM scan. One block per direction, 8 waves.
// LDS: x_t bf16[64][128] | h bf16[64][128] | gates f32[64][512] | Whh bf16[512][128]
//      = 16K + 16K + 128K + 128K = 288 KB (within 320 KB WGP LDS).
// x_t staged per step by the Tensor Data Mover; Whh staged once (reused 512x).
// ---------------------------------------------------------------------------
__global__ void encoder_scan_kernel(const bf16* __restrict__ emb,
                                    const bf16* Wih_f, const bf16* Whh_f,
                                    const bf16* Wih_b, const bf16* Whh_b,
                                    bf16* __restrict__ enc, bf16* h_dec, float* c_dec) {
  const int dir = blockIdx.x;
  const bf16* Wih = dir ? Wih_b : Wih_f;
  const bf16* Whh = dir ? Whh_b : Whh_f;
  extern __shared__ char smem[];
  bf16*  xh    = (bf16*)smem;                   // 16 KB  (LDS offset 0: TDM target)
  bf16*  hs    = xh + 64 * 128;                 // 16 KB
  float* gates = (float*)(smem + 32768);        // 128 KB
  bf16*  whh   = (bf16*)(smem + 32768 + 131072);// 128 KB
  const int t0 = threadIdx.x, lane = t0 & 31, wave = t0 >> 5;

  // stage Whh into LDS once (int4-vectorized: 131072 B)
  {
    const int4* src = (const int4*)Whh;
    int4* dst = (int4*)whh;
    for (int i = t0; i < 8192; i += 256) dst[i] = src[i];
  }
  float creg[32];
#pragma unroll
  for (int i = 0; i < 32; ++i) creg[i] = 0.0f;
  for (int i = t0; i < 64 * 128; i += 256) hs[i] = (bf16)0.0f;
  __syncthreads();

  for (int step = 0; step < 512; ++step) {
    const int t = dir ? (511 - step) : step;
    // stage x_t = emb[:, t, :]  (64 rows x 128 bf16, row stride 512*128 elems)
#if USE_TDM
    if (wave == 0) {
      tdm_load_tile_bf16((uint64_t)(uintptr_t)(emb + (size_t)t * 128), 0u,
                         128, 64, 512 * 128);
      __builtin_amdgcn_s_wait_tensorcnt(0);
    }
#else
    for (int i = t0; i < 64 * 128; i += 256) {
      int b = i >> 7, e = i & 127;
      xh[i] = emb[((size_t)b * 512 + t) * 128 + e];
    }
#endif
    __syncthreads();
    // gates[64,512] = x_t @ Wih^T + h @ Whh^T   (WMMA, 128 tiles / 8 waves)
    for (int tile = wave; tile < 128; tile += 8) {
      int mt = tile >> 5, nt = tile & 31;
      v8f acc = {};
#pragma unroll
      for (int k0 = 0; k0 < 128; k0 += 32) {
        v16bf a = load_A_tile(xh, 128, mt * 16, k0, lane);
        v16bf b = load_B_tile_nk(Wih, 128, k0, nt * 16, lane);
        acc = WMMA_BF16(a, b, acc);
      }
#pragma unroll
      for (int k0 = 0; k0 < 128; k0 += 32) {
        v16bf a = load_A_tile(hs, 128, mt * 16, k0, lane);
        v16bf b = load_B_tile_nk(whh, 128, k0, nt * 16, lane);   // LDS-resident
        acc = WMMA_BF16(a, b, acc);
      }
      store_D_f32(gates, 512, mt * 16, nt * 16, lane, acc);
    }
    __syncthreads();
    // LSTM cell (gate order i,f,g,o)
#pragma unroll
    for (int i = 0; i < 32; ++i) {
      int p = i * 256 + t0;
      int b = p >> 7, j = p & 127;
      float ig = gates[b * 512 + j];
      float fg = gates[b * 512 + 128 + j];
      float gg = gates[b * 512 + 256 + j];
      float og = gates[b * 512 + 384 + j];
      float c  = sigf(fg) * creg[i] + sigf(ig) * tanhf(gg);
      creg[i]  = c;
      float h  = sigf(og) * tanhf(c);
      hs[b * 128 + j] = (bf16)h;
      enc[((size_t)b * 512 + t) * 256 + dir * 128 + j] = (bf16)h;
    }
    __syncthreads();
  }
#pragma unroll
  for (int i = 0; i < 32; ++i) {
    int p = i * 256 + t0;
    h_dec[dir * 64 * 128 + p] = hs[p];
    c_dec[dir * 64 * 128 + p] = creg[i];
  }
}

// ---------------------------------------------------------------------------
// Generic WMMA GEMM: Out[M,N] = A[M,K](bf16) @ Wnk[N,K]^T + bias
// ---------------------------------------------------------------------------
template <bool OUT_BF16>
__global__ void gemm_kernel(const bf16* __restrict__ A, const bf16* __restrict__ Wnk,
                            const float* __restrict__ bias, void* __restrict__ Out,
                            int M, int N, int K) {
  int lane = threadIdx.x & 31, wave = threadIdx.x >> 5;
  int tiles_n = N >> 4;
  int total   = (M >> 4) * tiles_n;
  int tile    = blockIdx.x * 8 + wave;
  if (tile >= total) return;      // wave-uniform; EXEC all-ones around WMMA
  int mt = tile / tiles_n, nt = tile % tiles_n;
  v8f acc = {};
  for (int k0 = 0; k0 < K; k0 += 32) {
    v16bf a = load_A_tile(A, K, mt * 16, k0, lane);
    v16bf b = load_B_tile_nk(Wnk, K, k0, nt * 16, lane);
    acc = WMMA_BF16(a, b, acc);
  }
  int n = lane & 15, mh = (lane < 16) ? 0 : 8;
  int col = nt * 16 + n;
  float bv = bias ? bias[col] : 0.0f;
#pragma unroll
  for (int r = 0; r < 8; ++r) {
    size_t idx = (size_t)(mt * 16 + mh + r) * N + col;
    float v = acc[r] + bv;
    if (OUT_BF16) ((bf16*)Out)[idx] = (bf16)v;
    else          ((float*)Out)[idx] = v;
  }
}

// ---------------------------------------------------------------------------
// Decoder cell: both dirs, fused WMMA GEMM + LSTM cell + _norm(dec).
// LDS: gates f32[64][512] | Whh bf16[512][128] (per-dir staged) |
//      decc bf16[64][256] | red f32[512]  = 290 KB.
// Min/max for _norm tracked on exact f32 values in registers.
// ---------------------------------------------------------------------------
__global__ void dec_cell_kernel(const bf16* __restrict__ dec_in, bf16* h_dec, float* c_dec,
                                const bf16* Wih_f, const bf16* Whh_f, const float* b_f,
                                const bf16* Wih_b, const bf16* Whh_b, const float* b_b,
                                bf16* __restrict__ dec_n) {
  extern __shared__ char smem[];
  float* gates = (float*)smem;                       // 128 KB
  bf16*  whh   = (bf16*)(smem + 131072);             // 128 KB (per-dir)
  bf16*  deccb = (bf16*)(smem + 262144);             // 32 KB
  float* red   = (float*)(smem + 262144 + 32768);    // 2 KB
  const int t0 = threadIdx.x, lane = t0 & 31, wave = t0 >> 5;
  float mn = 3.4e38f, mx = -3.4e38f;

  for (int dir = 0; dir < 2; ++dir) {
    const bf16* Wih  = dir ? Wih_b : Wih_f;
    const bf16* WhhG = dir ? Whh_b : Whh_f;
    const float* bia = dir ? b_b : b_f;
    bf16*  h = h_dec + dir * 64 * 128;
    float* c = c_dec + dir * 64 * 128;
    // stage this dir's Whh
    {
      const int4* src = (const int4*)WhhG;
      int4* dst = (int4*)whh;
      for (int i = t0; i < 8192; i += 256) dst[i] = src[i];
    }
    __syncthreads();
    // gates = dec_in @ Wih^T + h @ Whh^T + b
    for (int tile = wave; tile < 128; tile += 8) {
      int mt = tile >> 5, nt = tile & 31;
      v8f acc = {};
#pragma unroll
      for (int k0 = 0; k0 < 256; k0 += 32) {
        v16bf a = load_A_tile(dec_in, 256, mt * 16, k0, lane);
        v16bf b = load_B_tile_nk(Wih, 256, k0, nt * 16, lane);
        acc = WMMA_BF16(a, b, acc);
      }
#pragma unroll
      for (int k0 = 0; k0 < 128; k0 += 32) {
        v16bf a = load_A_tile(h, 128, mt * 16, k0, lane);
        v16bf b = load_B_tile_nk(whh, 128, k0, nt * 16, lane);   // LDS-resident
        acc = WMMA_BF16(a, b, acc);
      }
      int n = lane & 15, mh = (lane < 16) ? 0 : 8;
      int col = nt * 16 + n;
      float bv = bia[col];
#pragma unroll
      for (int r = 0; r < 8; ++r) gates[(mt * 16 + mh + r) * 512 + col] = acc[r] + bv;
    }
    __syncthreads();
    for (int i = t0; i < 64 * 128; i += 256) {
      int b = i >> 7, j = i & 127;
      float ig = gates[b * 512 + j];
      float fg = gates[b * 512 + 128 + j];
      float gg = gates[b * 512 + 256 + j];
      float og = gates[b * 512 + 384 + j];
      float cn = sigf(fg) * c[i] + sigf(ig) * tanhf(gg);
      c[i] = cn;
      float hv = sigf(og) * tanhf(cn);
      h[i] = (bf16)hv;
      deccb[b * 256 + dir * 128 + j] = (bf16)hv;
      mn = fminf(mn, hv); mx = fmaxf(mx, hv);
    }
    __syncthreads();
  }
  // dec = _norm(concat(hf2, hb2)) over the whole [64,256] tensor
  red[t0] = mn; red[256 + t0] = mx; __syncthreads();
  for (int s = 128; s > 0; s >>= 1) {
    if (t0 < s) { red[t0] = fminf(red[t0], red[t0 + s]); red[256 + t0] = fmaxf(red[256 + t0], red[256 + t0 + s]); }
    __syncthreads();
  }
  mn = red[0]; mx = red[256];
  float inv = 1.0f / (mx - mn);
  for (int i = t0; i < 64 * 256; i += 256) dec_n[i] = (bf16)(((float)deccb[i] - mn) * inv);
}

// ---------------------------------------------------------------------------
// Attention scores: ow/aw logits for all (b,s). One block = 256 s of one b.
// ---------------------------------------------------------------------------
__global__ void scores_kernel(const bf16* __restrict__ w1e, const bf16* __restrict__ w2e,
                              const float* __restrict__ da1, const float* __restrict__ da2,
                              const float* __restrict__ v1W, const float* __restrict__ v1b,
                              const float* __restrict__ v2W, const float* __restrict__ v2b,
                              const float* __restrict__ s_inputs, const float* __restrict__ v_input,
                              const float* __restrict__ actions,
                              float* __restrict__ ows, float* __restrict__ awlog, int step) {
  __shared__ float sda1[256], sda2[256], sv1[256], sv2[256];
  const int t0 = threadIdx.x;
  const int b  = blockIdx.x >> 1;
  const int s  = ((blockIdx.x & 1) << 8) + t0;
  sda1[t0] = da1[b * 256 + t0];
  sda2[t0] = da2[b * 256 + t0];
  sv1[t0]  = v1W[t0];
  sv2[t0]  = v2W[t0];
  __syncthreads();
  const bf16* w1 = w1e + ((size_t)b * 512 + s) * 256;
  const bf16* w2 = w2e + ((size_t)b * 512 + s) * 256;
  __builtin_prefetch(w1, 0, 0);
  __builtin_prefetch(w2, 0, 0);
  float o1 = 0.0f, o2 = 0.0f;
#pragma unroll 4
  for (int j = 0; j < 256; ++j) {
    o1 += tanhf((float)w1[j] + sda1[j]) * sv1[j];
    o2 += tanhf((float)w2[j] + sda2[j]) * sv2[j];
  }
  float sat    = (s_inputs[((size_t)b * 512 + s) * 3] < v_input[step * 3]) ? 1.0f : 0.0f;
  float cannot = sat + actions[b * 512 + s];
  ows[((size_t)step * 64 + b) * 512 + s] = o1 + v1b[0] - cannot * 1e-6f;
  awlog[b * 512 + s] = o2 + v2b[0];
}

// ---------------------------------------------------------------------------
// Per-batch: argmax(ow), softmax(aw), ctx = aw @ enc[b], action scatter.
// ---------------------------------------------------------------------------
__global__ void select_kernel(const float* __restrict__ ows, const float* __restrict__ awlog,
                              const bf16* __restrict__ enc, bf16* __restrict__ dec_in,
                              float* actions, int* sels, int step) {
  const int b = blockIdx.x, t0 = threadIdx.x;
  __shared__ float p[512];
  __shared__ float red[256];
  __shared__ int   redi[256];
  const float* ow = ows + ((size_t)step * 64 + b) * 512;
  const float* al = awlog + b * 512;

  float bv = -3.4e38f; int bi = 0x7fffffff;
  for (int s = t0; s < 512; s += 256) {
    float v = ow[s];
    if (v > bv || (v == bv && s < bi)) { bv = v; bi = s; }
  }
  red[t0] = bv; redi[t0] = bi; __syncthreads();
  for (int st = 128; st > 0; st >>= 1) {
    if (t0 < st) {
      if (red[t0 + st] > red[t0] || (red[t0 + st] == red[t0] && redi[t0 + st] < redi[t0])) {
        red[t0] = red[t0 + st]; redi[t0] = redi[t0 + st];
      }
    }
    __syncthreads();
  }
  int sel = redi[0];
  __syncthreads();

  float m = -3.4e38f;
  for (int s = t0; s < 512; s += 256) m = fmaxf(m, al[s]);
  red[t0] = m; __syncthreads();
  for (int st = 128; st > 0; st >>= 1) { if (t0 < st) red[t0] = fmaxf(red[t0], red[t0 + st]); __syncthreads(); }
  m = red[0]; __syncthreads();
  float sum = 0.0f;
  for (int s = t0; s < 512; s += 256) { float e = __expf(al[s] - m); p[s] = e; sum += e; }
  red[t0] = sum; __syncthreads();
  for (int st = 128; st > 0; st >>= 1) { if (t0 < st) red[t0] += red[t0 + st]; __syncthreads(); }
  float inv = 1.0f / red[0];
  __syncthreads();

  float acc = 0.0f;
  for (int s = 0; s < 512; ++s) acc += p[s] * (float)enc[((size_t)b * 512 + s) * 256 + t0];
  dec_in[b * 256 + t0] = (bf16)(acc * inv);

  if (t0 == 0) {
    sels[step * 64 + b] = sel;
    actions[b * 512 + sel] += 1.0f;
  }
}

__global__ void finalize_kernel(const int* __restrict__ sels, const int* __restrict__ s_node_indexes,
                                float* __restrict__ out) {
  int gid = blockIdx.x * 256 + threadIdx.x;   // 64*64
  int b = gid >> 6, v = gid & 63;
  int sel = sels[v * 64 + b];
  out[b * 64 + v]           = (float)s_node_indexes[b * 512 + sel];   // original
  out[64 * 64 + b * 64 + v] = (float)sel;                            // shuffled
}

// ---------------------------------------------------------------------------
// Host orchestration
// ---------------------------------------------------------------------------
static inline size_t align256(size_t x) { return (x + 255) & ~(size_t)255; }

extern "C" void kernel_launch(void* const* d_in, const int* in_sizes, int n_in,
                              void* d_out, int out_size, void* d_ws, size_t ws_size,
                              hipStream_t stream) {
  (void)in_sizes; (void)n_in; (void)out_size; (void)ws_size;
  const int B = 64, S = 512, V = 64, E = 128, H = 128;

  const int*   s_node_indexes = (const int*)d_in[0];
  const float* s_inputs  = (const float*)d_in[1];
  const float* v_input   = (const float*)d_in[2];
  const float* emb_W     = (const float*)d_in[3];
  const float* f_enc_Wih_f = (const float*)d_in[4];
  const float* f_enc_Whh_f = (const float*)d_in[5];
  const float* f_enc_Wih_b = (const float*)d_in[6];
  const float* f_enc_Whh_b = (const float*)d_in[7];
  const float* f_dec_Wih_f = (const float*)d_in[8];
  const float* f_dec_Whh_f = (const float*)d_in[9];
  const float* dec_b_f   = (const float*)d_in[10];
  const float* f_dec_Wih_b = (const float*)d_in[11];
  const float* f_dec_Whh_b = (const float*)d_in[12];
  const float* dec_b_b   = (const float*)d_in[13];
  const float* f_w1_W    = (const float*)d_in[14];
  const float* w1_b      = (const float*)d_in[15];
  const float* f_a1_W    = (const float*)d_in[16];
  const float* a1_b      = (const float*)d_in[17];
  const float* v1_W      = (const float*)d_in[18];
  const float* v1_b      = (const float*)d_in[19];
  const float* f_w2_W    = (const float*)d_in[20];
  const float* w2_b      = (const float*)d_in[21];
  const float* f_a2_W    = (const float*)d_in[22];
  const float* a2_b      = (const float*)d_in[23];
  const float* v2_W      = (const float*)d_in[24];
  const float* v2_b      = (const float*)d_in[25];

  char* ws = (char*)d_ws; size_t cur = 0;
  auto alloc = [&](size_t bytes) -> void* { void* p = ws + cur; cur = align256(cur + bytes); return p; };
  int*   minmax  = (int*)  alloc(2 * sizeof(int));
  bf16*  emb     = (bf16*) alloc((size_t)B * S * E * 2);
  bf16*  enc     = (bf16*) alloc((size_t)B * S * 2 * H * 2);
  bf16*  w1e     = (bf16*) alloc((size_t)B * S * 2 * H * 2);
  bf16*  w2e     = (bf16*) alloc((size_t)B * S * 2 * H * 2);
  bf16*  enc_Wih_f = (bf16*)alloc(512 * 128 * 2);
  bf16*  enc_Whh_f = (bf16*)alloc(512 * 128 * 2);
  bf16*  enc_Wih_b = (bf16*)alloc(512 * 128 * 2);
  bf16*  enc_Whh_b = (bf16*)alloc(512 * 128 * 2);
  bf16*  dec_Wih_f = (bf16*)alloc(512 * 256 * 2);
  bf16*  dec_Whh_f = (bf16*)alloc(512 * 128 * 2);
  bf16*  dec_Wih_b = (bf16*)alloc(512 * 256 * 2);
  bf16*  dec_Whh_b = (bf16*)alloc(512 * 128 * 2);
  bf16*  w1W = (bf16*)alloc(256 * 256 * 2);
  bf16*  a1W = (bf16*)alloc(256 * 256 * 2);
  bf16*  w2W = (bf16*)alloc(256 * 256 * 2);
  bf16*  a2W = (bf16*)alloc(256 * 256 * 2);
  bf16*  h_dec  = (bf16*) alloc(2 * 64 * 128 * 2);
  float* c_dec  = (float*)alloc(2 * 64 * 128 * 4);
  bf16*  dec_in = (bf16*) alloc(64 * 256 * 2);
  bf16*  dec_n  = (bf16*) alloc(64 * 256 * 2);
  float* da1    = (float*)alloc(64 * 256 * 4);
  float* da2    = (float*)alloc(64 * 256 * 4);
  float* awlog  = (float*)alloc(64 * 512 * 4);
  float* actions= (float*)alloc(64 * 512 * 4);
  int*   sels   = (int*)  alloc(64 * 64 * 4);

  float* out_f = (float*)d_out;
  float* ows   = out_f + 2 * B * V;   // [V,B,S] written directly into d_out

  init_kernel<<<128, 256, 0, stream>>>(minmax, actions, dec_in);

  struct { const float* s; bf16* d; int n; } cv[] = {
    {f_enc_Wih_f, enc_Wih_f, 512 * 128}, {f_enc_Whh_f, enc_Whh_f, 512 * 128},
    {f_enc_Wih_b, enc_Wih_b, 512 * 128}, {f_enc_Whh_b, enc_Whh_b, 512 * 128},
    {f_dec_Wih_f, dec_Wih_f, 512 * 256}, {f_dec_Whh_f, dec_Whh_f, 512 * 128},
    {f_dec_Wih_b, dec_Wih_b, 512 * 256}, {f_dec_Whh_b, dec_Whh_b, 512 * 128},
    {f_w1_W, w1W, 256 * 256}, {f_a1_W, a1W, 256 * 256},
    {f_w2_W, w2W, 256 * 256}, {f_a2_W, a2W, 256 * 256},
  };
  for (auto& c : cv)
    cvt_bf16_kernel<<<(c.n + 255) / 256, 256, 0, stream>>>(c.s, c.d, c.n);

  emb_kernel<<<(B * S * E) / 256, 256, 0, stream>>>(s_inputs, emb_W, emb, minmax);
  norm_emb_kernel<<<(B * S * E) / 256, 256, 0, stream>>>(emb, minmax);

  size_t enc_lds = 32768 + 131072 + 131072;   // 288 KB
  encoder_scan_kernel<<<2, 256, enc_lds, stream>>>(emb, enc_Wih_f, enc_Whh_f,
                                                   enc_Wih_b, enc_Whh_b, enc, h_dec, c_dec);

  {
    int M = B * S, N = 2 * H, K = 2 * H;
    int blocks = ((M / 16) * (N / 16) + 7) / 8;
    gemm_kernel<true><<<blocks, 256, 0, stream>>>(enc, w1W, w1_b, w1e, M, N, K);
    gemm_kernel<true><<<blocks, 256, 0, stream>>>(enc, w2W, w2_b, w2e, M, N, K);
  }

  size_t dec_lds = 131072 + 131072 + 32768 + 2048;   // 290 KB
  for (int step = 0; step < V; ++step) {
    dec_cell_kernel<<<1, 256, dec_lds, stream>>>(dec_in, h_dec, c_dec,
                                                 dec_Wih_f, dec_Whh_f, dec_b_f,
                                                 dec_Wih_b, dec_Whh_b, dec_b_b, dec_n);
    {
      int M = 64, N = 256, K = 256;
      int blocks = ((M / 16) * (N / 16) + 7) / 8;
      gemm_kernel<false><<<blocks, 256, 0, stream>>>(dec_n, a1W, a1_b, da1, M, N, K);
      gemm_kernel<false><<<blocks, 256, 0, stream>>>(dec_n, a2W, a2_b, da2, M, N, K);
    }
    scores_kernel<<<(B * S) / 256, 256, 0, stream>>>(w1e, w2e, da1, da2, v1_W, v1_b, v2_W, v2_b,
                                                     s_inputs, v_input, actions, ows, awlog, step);
    select_kernel<<<B, 256, 0, stream>>>(ows, awlog, enc, dec_in, actions, sels, step);
  }

  finalize_kernel<<<(B * V) / 256, 256, 0, stream>>>(sels, s_node_indexes, out_f);
}
